// Decoder_3126736191834
// MI455X (gfx1250) — compile-verified, matrix operand order checked by reading, not run on hardware
//
#include <hip/hip_runtime.h>

// ---------------------------------------------------------------------------
// CDNA5 (gfx1250) decoder block.
// - All matmuls: v_wmma_f32_16x16x32_bf16, f32 accumulation.
// - Operands pre-converted to bf16 once (producers emit bf16; weights get a
//   one-time convert+transpose) -> GEMM staging is a pure byte move done with
//   global_load_async_to_lds_b128 (ASYNCcnt) into double-buffered LDS, so the
//   copies of tile k+1 overlap the 8 WMMAs on tile k.
// - Streaming fp32 tensors (attn2, final output) use non-temporal TH hints so
//   the 512MB attn2 stream does not wash the 192MB L2.
// ---------------------------------------------------------------------------

typedef __attribute__((ext_vector_type(16))) __bf16 v16bf;
typedef __attribute__((ext_vector_type(8)))  float  v8f;

#define TILE_M 128
#define TILE_N 128
#define TILE_K 32
#define LDS_PAD 40  // ushorts/row: 80B stride = 20-bank stride (conflict-free), 16B aligned

__device__ __forceinline__ unsigned short f2bf(float f) {
  union { float f; unsigned int u; } v; v.f = f;
  unsigned int r = v.u + 0x7FFFu + ((v.u >> 16) & 1u);  // RNE
  return (unsigned short)(r >> 16);
}
__device__ __forceinline__ float bf2f(unsigned short u) {
  union { unsigned int u; float f; } v; v.u = (unsigned int)u << 16;
  return v.f;
}

union FragU { v16bf v; unsigned int u[8]; };

// Gather one 16x32 bf16 WMMA operand fragment from an LDS tile [row][k]
// (stride LDS_PAD). ISA 16-bit A-matrix layout: lane L holds row (L&15),
// K-octet base kb=(L>>4)*8; VGPR j = pair {kb+2j}, VGPR j+4 = {16+kb+2j}.
// Compiler merges these into two ds_load_b128 per fragment.
__device__ __forceinline__ v16bf load_frag(const unsigned short* tile, int row0, int lane) {
  const unsigned short* p = tile + (row0 + (lane & 15)) * LDS_PAD;
  const int kb = (lane >> 4) << 3;
  FragU f;
#pragma unroll
  for (int j = 0; j < 4; ++j) {
    f.u[j]     = *(const unsigned int*)(p + kb + 2 * j);
    f.u[j + 4] = *(const unsigned int*)(p + 16 + kb + 2 * j);
  }
  return f.v;
}

// Issue async copies of one 128x32 bf16 A-tile and 128x32 bf16 B-tile into LDS.
// Each thread moves 2x16B per tile. Out-of-range rows are CLAMPED, not
// branched: the duplicate data lands only in C rows/cols that the epilogue
// masks anyway, and we avoid saveexec/branch ladders around each async op.
__device__ __forceinline__ void stage_async(unsigned short* dA, unsigned short* dB,
                                            const unsigned short* Ab, int lda,
                                            const unsigned short* Bb, int ldb,
                                            int tileM, int tileN, int M, int N,
                                            int k0, int tid)
{
#pragma unroll
  for (int it = 0; it < 2; ++it) {
    int chunk = it * 256 + tid;          // 0..511
    int r = chunk >> 2;                  // 0..127
    int c = (chunk & 3) << 3;            // bf16 col 0,8,16,24
    {
      int gr = tileM + r; gr = (gr < M) ? gr : (M - 1);
      unsigned loff = (unsigned)(unsigned long long)(const void*)(dA + r * LDS_PAD + c);
      unsigned long long gp =
          (unsigned long long)(const void*)(Ab + (long long)gr * lda + k0 + c);
      asm volatile("global_load_async_to_lds_b128 %0, %1, off"
                   :: "v"(loff), "v"(gp) : "memory");
    }
    {
      int gn = tileN + r; gn = (gn < N) ? gn : (N - 1);
      unsigned loff = (unsigned)(unsigned long long)(const void*)(dB + r * LDS_PAD + c);
      unsigned long long gp =
          (unsigned long long)(const void*)(Bb + (long long)gn * ldb + k0 + c);
      asm volatile("global_load_async_to_lds_b128 %0, %1, off"
                   :: "v"(loff), "v"(gp) : "memory");
    }
  }
}

// Batched GEMM, bf16 operands, both row-major over K:
//   C[b][m][n] = sum_k A[b][m][k] * B[b][n][k]  (+ bias[n])
// batch b = b1*nb2 + b2 (two-level strides serve per-n and per-(n,h) GEMMs).
// Writes fp32 C (Cf, non-temporal) and/or bf16 C (Cbf) — whichever non-null.
extern "C" __global__ void __launch_bounds__(256)
gemm_bf16_wmma(const unsigned short* __restrict__ A, long long sA1, long long sA2, int lda,
               const unsigned short* __restrict__ B, long long sB1, long long sB2, int ldb,
               const float* __restrict__ bias,
               float* __restrict__ Cf, unsigned short* __restrict__ Cbf,
               long long sC1, long long sC2, int ldc,
               int M, int N, int K, int nb2, float alpha)
{
  __shared__ unsigned short Asm[2][TILE_M * LDS_PAD];
  __shared__ unsigned short Bsm[2][TILE_N * LDS_PAD];

  const int tid  = threadIdx.x;
  const int lane = tid & 31;
  const int wave = tid >> 5;              // 8 wave32 waves
  const int rowBase = (wave >> 2) * 64;   // 2 wave-rows x 4 wave-cols
  const int colBase = (wave & 3) * 32;

  const int tileM = blockIdx.y * TILE_M;
  const int tileN = blockIdx.x * TILE_N;
  const int b  = blockIdx.z;
  const int b1 = b / nb2;
  const int b2 = b - b1 * nb2;

  const unsigned short* Ab = A + b1 * sA1 + b2 * sA2;
  const unsigned short* Bb = B + b1 * sB1 + b2 * sB2;

  v8f acc[4][2];
  v8f zero = {};
#pragma unroll
  for (int mi = 0; mi < 4; ++mi)
#pragma unroll
    for (int ni = 0; ni < 2; ++ni) acc[mi][ni] = zero;

  const int nsteps = K / TILE_K;  // all K here are multiples of 32
  stage_async(Asm[0], Bsm[0], Ab, lda, Bb, ldb, tileM, tileN, M, N, 0, tid);

  for (int s = 0; s < nsteps; ++s) {
    const int cur = s & 1;
    asm volatile("s_wait_asynccnt 0" ::: "memory");
    __syncthreads();  // buf[cur] staged; everyone done reading buf[cur^1]
    if (s + 1 < nsteps)
      stage_async(Asm[cur ^ 1], Bsm[cur ^ 1], Ab, lda, Bb, ldb,
                  tileM, tileN, M, N, (s + 1) * TILE_K, tid);

    v16bf bfr[2];
#pragma unroll
    for (int ni = 0; ni < 2; ++ni)
      bfr[ni] = load_frag(Bsm[cur], colBase + ni * 16, lane);
#pragma unroll
    for (int mi = 0; mi < 4; ++mi) {
      v16bf af = load_frag(Asm[cur], rowBase + mi * 16, lane);
#pragma unroll
      for (int ni = 0; ni < 2; ++ni)
        acc[mi][ni] = __builtin_amdgcn_wmma_f32_16x16x32_bf16(
            false, af, false, bfr[ni], (short)0, acc[mi][ni], false, false);
    }
  }

  // C/D f32 layout: vgpr r -> M = r + 8*(lane>>4), N = lane&15.
  float* CfB = Cf ? Cf + b1 * sC1 + b2 * sC2 : nullptr;
  unsigned short* CbB = Cbf ? Cbf + b1 * sC1 + b2 * sC2 : nullptr;
  const int half = lane >> 4;
  const int ncol = lane & 15;
#pragma unroll
  for (int ni = 0; ni < 2; ++ni) {
    int gcol = tileN + colBase + ni * 16 + ncol;
    if (gcol >= N) continue;
    float bv = bias ? bias[gcol] : 0.f;
#pragma unroll
    for (int mi = 0; mi < 4; ++mi) {
#pragma unroll
      for (int r = 0; r < 8; ++r) {
        int grow = tileM + rowBase + mi * 16 + half * 8 + r;
        if (grow < M) {
          long long off = (long long)grow * ldc + gcol;
          float val = acc[mi][ni][r] * alpha + bv;
          if (CfB) __builtin_nontemporal_store(val, &CfB[off]);  // streaming fp32
          if (CbB) CbB[off] = f2bf(val);
        }
      }
    }
  }
}

// Elementwise fp32 -> bf16.
extern "C" __global__ void __launch_bounds__(256)
cvt_f32_bf16(const float* __restrict__ src, unsigned short* __restrict__ dst, long long n)
{
  long long i = ((long long)blockIdx.x * 256 + threadIdx.x) * 4;
  if (i + 3 < n) {
    float4 v = *(const float4*)(src + i);
    unsigned int lo = (unsigned int)f2bf(v.x) | ((unsigned int)f2bf(v.y) << 16);
    unsigned int hi = (unsigned int)f2bf(v.z) | ((unsigned int)f2bf(v.w) << 16);
    unsigned int* d = (unsigned int*)(dst + i);
    d[0] = lo; d[1] = hi;
  } else {
    for (int j = 0; j < 4; ++j)
      if (i + j < n) dst[i + j] = f2bf(src[i + j]);
  }
}

// Batched transpose (+optional fp32->bf16): dst[c][r] = src[r][c].
extern "C" __global__ void __launch_bounds__(256)
transpose_cvt_bf16(const void* __restrict__ src, int srcIsBf16,
                   long long sS1, long long sS2, int lds_,
                   unsigned short* __restrict__ dst,
                   long long sD1, long long sD2, int ldd,
                   int rows, int cols, int nb2)
{
  __shared__ unsigned short t[32][33];
  const int b = blockIdx.z, b1 = b / nb2, b2 = b - b1 * nb2;
  const float* sf = (const float*)src;
  const unsigned short* sb = (const unsigned short*)src;
  const long long sbase = b1 * sS1 + b2 * sS2;
  const int r0 = blockIdx.y * 32, c0 = blockIdx.x * 32;
  const int tx = threadIdx.x & 31, ty = threadIdx.x >> 5;
#pragma unroll
  for (int j = 0; j < 4; ++j) {
    int r = r0 + ty + j * 8, c = c0 + tx;
    unsigned short v = 0;
    if (r < rows && c < cols) {
      long long idx = sbase + (long long)r * lds_ + c;
      v = srcIsBf16 ? sb[idx] : f2bf(sf[idx]);
    }
    t[tx][ty + j * 8] = v;
  }
  __syncthreads();
  const long long dbase = b1 * sD1 + b2 * sD2;
#pragma unroll
  for (int j = 0; j < 4; ++j) {
    int c = c0 + ty + j * 8, r = r0 + tx;
    if (c < cols && r < rows)
      dst[dbase + (long long)c * ldd + r] = t[ty + j * 8][tx];
  }
}

// Row softmax, fp32 or bf16 source, fp32 and/or bf16 destinations.
// Reference semantics: masked -> -1e9, then *scale, then softmax. Lk == 1024.
// fp32 path (the 512MB attn2 stream) uses non-temporal load/store.
extern "C" __global__ void __launch_bounds__(256)
softmax_rows(const float* __restrict__ srcF, const unsigned short* __restrict__ srcB,
             float* __restrict__ dstF, unsigned short* __restrict__ dstB,
             const int* __restrict__ mask, int H, int L, int Lk, float scale)
{
  const int q  = blockIdx.x;
  const int bh = blockIdx.y;
  const int n  = bh / H;
  const long long rb = ((long long)bh * L + q) * (long long)Lk;
  const int* mrow = mask ? mask + ((long long)n * L + q) * (long long)Lk : nullptr;
  const int tid = threadIdx.x;

  float v[4];
  float mx = -3.4e38f;
#pragma unroll
  for (int j = 0; j < 4; ++j) {
    int k = j * 256 + tid;
    float s = srcB ? bf2f(srcB[rb + k]) : __builtin_nontemporal_load(&srcF[rb + k]);
    if (mrow && mrow[k] == 0) s = -1e9f;
    s *= scale;
    v[j] = s;
    mx = fmaxf(mx, s);
  }
#pragma unroll
  for (int off = 16; off > 0; off >>= 1) mx = fmaxf(mx, __shfl_xor(mx, off));
  __shared__ float redm[8], reds[8];
  if ((tid & 31) == 0) redm[tid >> 5] = mx;
  __syncthreads();
  mx = redm[0];
#pragma unroll
  for (int w = 1; w < 8; ++w) mx = fmaxf(mx, redm[w]);

  float sum = 0.f;
#pragma unroll
  for (int j = 0; j < 4; ++j) { v[j] = __expf(v[j] - mx); sum += v[j]; }
#pragma unroll
  for (int off = 16; off > 0; off >>= 1) sum += __shfl_xor(sum, off);
  if ((tid & 31) == 0) reds[tid >> 5] = sum;
  __syncthreads();
  sum = 0.f;
#pragma unroll
  for (int w = 0; w < 8; ++w) sum += reds[w];
  const float inv = 1.0f / sum;
#pragma unroll
  for (int j = 0; j < 4; ++j) {
    int k = j * 256 + tid;
    float o = v[j] * inv;
    if (dstF) __builtin_nontemporal_store(o, &dstF[rb + k]);
    if (dstB) dstB[rb + k] = f2bf(o);
  }
}

// ---------------------------------------------------------------------------
extern "C" void kernel_launch(void* const* d_in, const int* in_sizes, int n_in,
                              void* d_out, int out_size, void* d_ws, size_t ws_size,
                              hipStream_t stream)
{
  (void)in_sizes; (void)n_in; (void)out_size; (void)ws_size;
  typedef unsigned short u16;
  typedef long long ll;

  const float* data  = (const float*)d_in[0];
  const float* input = (const float*)d_in[1];
  const int*   mask  = (const int*)d_in[2];
  const float* w_qkv = (const float*)d_in[3];
  const float* b_qkv = (const float*)d_in[4];
  const float* w_fc1 = (const float*)d_in[5];
  const float* b_fc1 = (const float*)d_in[6];
  const float* w_kv  = (const float*)d_in[7];
  const float* b_kv  = (const float*)d_in[8];
  const float* w_q   = (const float*)d_in[9];
  const float* b_q   = (const float*)d_in[10];
  const float* w_fc2 = (const float*)d_in[11];
  const float* b_fc2 = (const float*)d_in[12];

  constexpr int Nb = 8, L = 1024, DIN = 1024, H = 16, HD = 64, DIM = 1024, FF = 1024;
  const float is = 0.125f;  // 1/sqrt(HD)

  float* out_main = (float*)d_out;                       // [8,1024,1024]
  float* attn2    = out_main + (ll)Nb * L * FF;          // [8,16,1024,1024] fp32 output

  // --- workspace carve (bf16 = u16), with safe aliasing -------------------
  char* p = (char*)d_ws;
  auto take = [&p](size_t bytes) { char* r = p; p += (bytes + 255) & ~(size_t)255; return r; };
  u16* scores_bf = (u16*)take((size_t)Nb * H * L * L * 2);   // 256MB: self scores/attn
  u16* qkv_bf    = (u16*)take((size_t)Nb * L * 3 * DIM * 2); // 48MB
  u16* bf_input  = (u16*)take((size_t)Nb * L * DIN * 2);
  u16* bf_data   = (u16*)take((size_t)Nb * L * DIN * 2);
  u16* wqkvT     = (u16*)take((size_t)DIN * 3 * DIM * 2);
  u16* wfc1T     = (u16*)take((size_t)DIM * DIM * 2);
  u16* wkvT      = (u16*)take((size_t)DIN * 2 * DIM * 2);
  u16* wqT       = (u16*)take((size_t)DIM * DIM * 2);
  u16* wfc2T     = (u16*)take((size_t)DIM * FF * 2);
  u16* vT_bf     = (u16*)take((size_t)Nb * H * HD * L * 2);  // 16MB
  u16* out_bf    = (u16*)take((size_t)Nb * L * DIM * 2);
  u16* x_bf      = (u16*)take((size_t)Nb * L * DIM * 2);
  // aliases (regions dead by the time these are written):
  u16* kv_bf     = qkv_bf;     // qkv dead after stage 2/1b; kv written stage 6
  u16* attn2_bf  = scores_bf;  // self attn dead after stage 4; written stage 9
  u16* v2T_bf    = vT_bf;      // vT dead after stage 4; written stage 6b
  u16* q2_bf     = out_bf;     // out dead after stage 5; written stage 7
  u16* out2_bf   = x_bf;       // x dead after stage 7; written stage 10

  dim3 blk(256);
  auto g  = [](int n_, int m_, int b_) { return dim3((n_ + 127) / 128, (m_ + 127) / 128, b_); };
  auto gt = [](int c_, int r_, int b_) { return dim3((c_ + 31) / 32, (r_ + 31) / 32, b_); };
  auto gc = [](ll n_) { return dim3((unsigned)((n_ + 1023) / 1024)); };

  // --- one-time operand preparation ---------------------------------------
  cvt_f32_bf16<<<gc((ll)Nb * L * DIN), blk, 0, stream>>>(input, bf_input, (ll)Nb * L * DIN);
  cvt_f32_bf16<<<gc((ll)Nb * L * DIN), blk, 0, stream>>>(data, bf_data, (ll)Nb * L * DIN);
  transpose_cvt_bf16<<<gt(3 * DIM, DIN, 1), blk, 0, stream>>>(w_qkv, 0, 0, 0, 3 * DIM, wqkvT, 0, 0, DIN, DIN, 3 * DIM, 1);
  transpose_cvt_bf16<<<gt(DIM, DIM, 1), blk, 0, stream>>>(w_fc1, 0, 0, 0, DIM, wfc1T, 0, 0, DIM, DIM, DIM, 1);
  transpose_cvt_bf16<<<gt(2 * DIM, DIN, 1), blk, 0, stream>>>(w_kv, 0, 0, 0, 2 * DIM, wkvT, 0, 0, DIN, DIN, 2 * DIM, 1);
  transpose_cvt_bf16<<<gt(DIM, DIM, 1), blk, 0, stream>>>(w_q, 0, 0, 0, DIM, wqT, 0, 0, DIM, DIM, DIM, 1);
  transpose_cvt_bf16<<<gt(FF, DIM, 1), blk, 0, stream>>>(w_fc2, 0, 0, 0, FF, wfc2T, 0, 0, DIM, DIM, FF, 1);

  // 1. qkv = input @ w_qkv + b_qkv   -> bf16 [8,L,3072]
  gemm_bf16_wmma<<<g(3 * DIM, L, Nb), blk, 0, stream>>>(
      bf_input, (ll)L * DIN, 0, DIN, wqkvT, 0, 0, DIN, b_qkv,
      nullptr, qkv_bf, (ll)L * 3 * DIM, 0, 3 * DIM, L, 3 * DIM, DIN, 1, 1.f);

  // 1b. vT[n,h] = V^T  (bf16 [HD,L] per head)
  transpose_cvt_bf16<<<gt(HD, L, Nb * H), blk, 0, stream>>>(
      qkv_bf + 2 * DIM, 1, (ll)L * 3 * DIM, HD, 3 * DIM,
      vT_bf, (ll)H * HD * L, (ll)HD * L, L, L, HD, H);

  // 2. scores[n,h] = q @ k^T   -> bf16
  gemm_bf16_wmma<<<g(L, L, Nb * H), blk, 0, stream>>>(
      qkv_bf, (ll)L * 3 * DIM, HD, 3 * DIM,
      qkv_bf + DIM, (ll)L * 3 * DIM, HD, 3 * DIM, nullptr,
      nullptr, scores_bf, (ll)H * L * L, (ll)L * L, L, L, L, HD, H, 1.f);

  // 3. attn = softmax(mask(scores)/sqrt(d))  (bf16 in place)
  softmax_rows<<<dim3(L, Nb * H), blk, 0, stream>>>(
      nullptr, scores_bf, nullptr, scores_bf, mask, H, L, L, is);

  // 4. out = attn @ v   -> bf16 [8,L,DIM]
  gemm_bf16_wmma<<<g(HD, L, Nb * H), blk, 0, stream>>>(
      scores_bf, (ll)H * L * L, (ll)L * L, L,
      vT_bf, (ll)H * HD * L, (ll)HD * L, L, nullptr,
      nullptr, out_bf, (ll)L * DIM, HD, DIM, L, HD, L, H, 1.f);

  // 5. x = out @ w_fc1 + b_fc1  -> bf16
  gemm_bf16_wmma<<<g(DIM, L, Nb), blk, 0, stream>>>(
      out_bf, (ll)L * DIM, 0, DIM, wfc1T, 0, 0, DIM, b_fc1,
      nullptr, x_bf, (ll)L * DIM, 0, DIM, L, DIM, DIM, 1, 1.f);

  // 6. kv = data @ w_kv + b_kv  -> bf16 [8,L,2048]
  gemm_bf16_wmma<<<g(2 * DIM, L, Nb), blk, 0, stream>>>(
      bf_data, (ll)L * DIN, 0, DIN, wkvT, 0, 0, DIN, b_kv,
      nullptr, kv_bf, (ll)L * 2 * DIM, 0, 2 * DIM, L, 2 * DIM, DIN, 1, 1.f);

  // 6b. v2T[n,h] = V2^T
  transpose_cvt_bf16<<<gt(HD, L, Nb * H), blk, 0, stream>>>(
      kv_bf + DIM, 1, (ll)L * 2 * DIM, HD, 2 * DIM,
      v2T_bf, (ll)H * HD * L, (ll)HD * L, L, L, HD, H);

  // 7. q2 = x @ w_q + b_q  -> bf16
  gemm_bf16_wmma<<<g(DIM, L, Nb), blk, 0, stream>>>(
      x_bf, (ll)L * DIM, 0, DIM, wqT, 0, 0, DIM, b_q,
      nullptr, q2_bf, (ll)L * DIM, 0, DIM, L, DIM, DIM, 1, 1.f);

  // 8. scores2 = q2 @ k2^T  -> fp32 (non-temporal) DIRECTLY into attn2 output
  gemm_bf16_wmma<<<g(L, L, Nb * H), blk, 0, stream>>>(
      q2_bf, (ll)L * DIM, HD, DIM,
      kv_bf, (ll)L * 2 * DIM, HD, 2 * DIM, nullptr,
      attn2, nullptr, (ll)H * L * L, (ll)L * L, L, L, L, HD, H, 1.f);

  // 9. attn2 = softmax(scores2/sqrt(d)): fp32 in place (output) + bf16 copy
  softmax_rows<<<dim3(L, Nb * H), blk, 0, stream>>>(
      attn2, nullptr, attn2, attn2_bf, nullptr, H, L, L, is);

  // 10. out2 = attn2 @ v2  -> bf16
  gemm_bf16_wmma<<<g(HD, L, Nb * H), blk, 0, stream>>>(
      attn2_bf, (ll)H * L * L, (ll)L * L, L,
      v2T_bf, (ll)H * HD * L, (ll)HD * L, L, nullptr,
      nullptr, out2_bf, (ll)L * DIM, HD, DIM, L, HD, L, H, 1.f);

  // 11. output = out2 @ w_fc2 + b_fc2  -> fp32 first segment of d_out
  gemm_bf16_wmma<<<g(FF, L, Nb), blk, 0, stream>>>(
      out2_bf, (ll)L * DIM, 0, DIM, wfc2T, 0, 0, FF, b_fc2,
      out_main, nullptr, (ll)L * FF, 0, FF, L, FF, DIM, 1, 1.f);
}
